// LRNN_25056839204869
// MI455X (gfx1250) — matrix operand order repaired and checked by reading, not compiled, not run on hardware
//
#include <hip/hip_runtime.h>
#include <hip/hip_bf16.h>
#include <math.h>

typedef __attribute__((ext_vector_type(16))) _Float16 v16h;
typedef __attribute__((ext_vector_type(8)))  _Float16 v8h;
typedef __attribute__((ext_vector_type(8)))  float    v8f;

// ---------------------------------------------------------------------------
// Zero-pad + f32->f16 convert: (N,C,H,W) f32 -> (N,C,H+2P,W+2P) f16.
// ---------------------------------------------------------------------------
__global__ void pad_f16_kernel(const float* __restrict__ in, _Float16* __restrict__ out,
                               int N, int C, int H, int W, int P, int total)
{
    int t = blockIdx.x * blockDim.x + threadIdx.x;
    if (t >= total) return;
    int Wp = W + 2 * P, Hp = H + 2 * P;
    int x = t % Wp;
    int y = (t / Wp) % Hp;
    int c = (t / (Wp * Hp)) % C;
    int n = t / (Wp * Hp * C);
    int ix = x - P, iy = y - P;
    float v = 0.f;
    if (ix >= 0 && ix < W && iy >= 0 && iy < H)
        v = in[((size_t)(n * C + c) * H + iy) * W + ix];
    out[t] = (_Float16)v;
}

// ---------------------------------------------------------------------------
// Implicit-GEMM convolution via CDNA5 WMMA (wave32, 16x16x32 f16 -> f32 acc).
// Input is pre-padded f16 -> all A-gather loads are unconditional full-wave
// loads from a wave-uniform base + per-lane (m + selected-const) offset.
// Each wave computes 16 pixels x (16*COT) output channels.
// ---------------------------------------------------------------------------
template<int KW, int ACT, int CIN, int COT>   // ACT: 0=none, 1=relu, 2=tanh
__global__ __launch_bounds__(128)
void conv_wmma_kernel(const _Float16* __restrict__ pin, const float* __restrict__ wgt,
                      const float* __restrict__ bias, float* __restrict__ out,
                      int N, int H, int W, int Cout, int numTiles)
{
    constexpr int KK   = KW * KW;
    constexpr int Ktot = CIN * KK;
    constexpr int Kpad = ((Ktot + 31) / 32) * 32;
    constexpr int PAD  = KW / 2;
    constexpr int NCO  = 16 * COT;

    extern __shared__ _Float16 wlds[];      // [NCO][Kpad], n-major, f16
    const int coBase = blockIdx.y * NCO;

    // Stage this cout-tile's weights into LDS (zero-pad K tail / co tail).
    for (int idx = threadIdx.x; idx < NCO * Kpad; idx += blockDim.x) {
        int n = idx / Kpad;
        int k = idx - n * Kpad;
        float v = 0.f;
        int co = coBase + n;
        if (k < Ktot && co < Cout) v = wgt[(size_t)co * Ktot + k];  // OIHW flat
        wlds[idx] = (_Float16)v;
    }
    __syncthreads();

    const int wave = threadIdx.x >> 5;
    const int lane = threadIdx.x & 31;
    const int tile = blockIdx.x * 4 + wave;
    if (tile >= numTiles) return;           // uniform per wave -> EXEC all-1s at WMMA

    const int m   = lane & 15;              // A row / B & D column index
    const int hig = lane >> 4;              // lane half (K-phase select)
    const int HW  = H * W;
    const int Wp   = W + 2 * PAD;
    const int HpWp = (H + 2 * PAD) * Wp;

    // 16 consecutive pixels in one row -> nImg, y, x0 are wave-uniform.
    const int p0   = tile * 16;
    const int nImg = p0 / HW;
    const int rem  = p0 - nImg * HW;
    const int y    = rem / W;
    const int x0   = rem - y * W;
    const _Float16* sBase = pin + (size_t)nImg * CIN * HpWp
                                + (size_t)(y + PAD) * Wp + (x0 + PAD);

    __builtin_prefetch(sBase, 0, 1);        // global_prefetch_b8

    const int kB0 = hig * 8;
    const _Float16* wrow = wlds + (size_t)m * Kpad;

    v8f acc[COT];
    #pragma unroll
    for (int j = 0; j < COT; ++j) acc[j] = (v8f){};

    #pragma unroll
    for (int kb = 0; kb < Kpad; kb += 32) {
        // ---- A fragment. Lane's k for element e: kb + 8*hig + (e<8 ? e : e+8).
        v16h a;
        #pragma unroll
        for (int e = 0; e < 16; ++e) {
            const int kL = kb + ((e < 8) ? e : (e + 8));        // compile-time
            const int kH = kL + 8;                              // compile-time
            const int ciL = kL / KK, rL = kL - ciL * KK;
            const int fyL = rL / KW, fxL = rL - fyL * KW;
            const int ciH = kH / KK, rH = kH - ciH * KK;
            const int fyH = rH / KW, fxH = rH - fyH * KW;
            const bool okL = (kL < Ktot), okH = (kH < Ktot);
            const int offL = ciL * HpWp + (fyL - PAD) * Wp + (fxL - PAD);
            const int offH = ciH * HpWp + (fyH - PAD) * Wp + (fxH - PAD);
            _Float16 v = (_Float16)0.f;
            if (okL && okH) {
                int o = hig ? offH : offL;                      // one v_cndmask
                v = sBase[o + m];                               // full-wave load
            } else if (okL) {
                if (hig == 0) v = sBase[offL + m];
            } else if (okH) {
                if (hig != 0) v = sBase[offH + m];
            }
            a[e] = v;
        }
        // ---- B fragments (one per cout tile), vectorized 16B LDS reads.
        #pragma unroll
        for (int j = 0; j < COT; ++j) {
            const _Float16* wr = wrow + (size_t)j * 16 * Kpad + kb + kB0;
            v8h b0 = *(const v8h*)(wr);
            v8h b1 = *(const v8h*)(wr + 16);
            v16h b;
            #pragma unroll
            for (int e = 0; e < 8; ++e) { b[e] = b0[e]; b[e + 8] = b1[e]; }
            acc[j] = __builtin_amdgcn_wmma_f32_16x16x32_f16(false, a, false, b,
                                                            (short)0, acc[j], false, false);
        }
    }

    // D layout: lane -> column N = lane&15; vgpr v -> row M = v + 8*hig
    #pragma unroll
    for (int j = 0; j < COT; ++j) {
        const int co = coBase + j * 16 + m;
        if (co < Cout) {
            const float bv = bias[co];
            #pragma unroll
            for (int v = 0; v < 8; ++v) {
                int pm = p0 + v + hig * 8;
                int nI = pm / HW;
                int r2 = pm - nI * HW;
                float d = acc[j][v] + bv;
                if (ACT == 1)      d = fmaxf(d, 0.f);
                else if (ACT == 2) d = tanhf(d);
                out[((size_t)nI * Cout + co) * HW + r2] = d;
            }
        }
    }
}

// ---------------------------------------------------------------------------
// Elementwise helpers
// ---------------------------------------------------------------------------
__global__ void maxpool_kernel(const float* __restrict__ in, float* __restrict__ out,
                               int N, int C, int H, int W, int total)
{
    int t = blockIdx.x * blockDim.x + threadIdx.x;
    if (t >= total) return;
    int oW = W >> 1, oH = H >> 1;
    int x = t % oW;
    int y = (t / oW) % oH;
    int c = (t / (oW * oH)) % C;
    int n = t / (oW * oH * C);
    const float* p = in + ((size_t)(n * C + c) * H + 2 * y) * W + 2 * x;
    float v = fmaxf(fmaxf(p[0], p[1]), fmaxf(p[W], p[W + 1]));
    out[((size_t)(n * C + c) * oH + y) * oW + x] = v;
}

__global__ void upsample_kernel(const float* __restrict__ in, int sC, int sOff,
                                float* __restrict__ out, int dC, int dOff,
                                int N, int C, int Hin, int Win, int s, int total)
{
    int t = blockIdx.x * blockDim.x + threadIdx.x;
    if (t >= total) return;
    int oW = Win * s, oH = Hin * s;
    int x = t % oW;
    int y = (t / oW) % oH;
    int c = (t / (oW * oH)) % C;
    int n = t / (oW * oH * C);
    float inv = 1.f / (float)s;
    float fy = (y + 0.5f) * inv - 0.5f;
    float fx = (x + 0.5f) * inv - 0.5f;
    int y0 = (int)floorf(fy);
    int x0 = (int)floorf(fx);
    float wy = fy - (float)y0;
    float wx = fx - (float)x0;
    int y0c = min(max(y0, 0), Hin - 1);
    int y1c = min(max(y0 + 1, 0), Hin - 1);
    int x0c = min(max(x0, 0), Win - 1);
    int x1c = min(max(x0 + 1, 0), Win - 1);
    const float* p = in + ((size_t)n * sC + sOff + c) * Hin * Win;
    float v00 = p[y0c * Win + x0c], v01 = p[y0c * Win + x1c];
    float v10 = p[y1c * Win + x0c], v11 = p[y1c * Win + x1c];
    float v = (1.f - wy) * ((1.f - wx) * v00 + wx * v01) +
              wy        * ((1.f - wx) * v10 + wx * v11);
    out[((size_t)n * dC + dOff + c) * oH * oW + y * oW + x] = v;
}

__global__ void copy_ch_kernel(const float* __restrict__ src, int sC, int sOff,
                               float* __restrict__ dst, int dC, int dOff,
                               int N, int C, int HW, int total)
{
    int t = blockIdx.x * blockDim.x + threadIdx.x;
    if (t >= total) return;
    int x = t % HW;
    int c = (t / HW) % C;
    int n = t / (HW * C);
    dst[((size_t)n * dC + dOff + c) * HW + x] = src[((size_t)n * sC + sOff + c) * HW + x];
}

// gate_v = concat([gy1 row0, gx1 rows 1..H-1], axis=2); gx1=c9[:,0:16], gy1=c9[:,16:32]
__global__ void gatev_kernel(const float* __restrict__ c9, float* __restrict__ gv,
                             int N, int H, int W, int total)
{
    int t = blockIdx.x * blockDim.x + threadIdx.x;
    if (t >= total) return;
    int x = t % W;
    int y = (t / W) % H;
    int c = (t / (W * H)) % 16;
    int n = t / (W * H * 16);
    int ch = ((y == 0) ? 16 : 0) + c;
    size_t HW = (size_t)H * W;
    gv[((size_t)n * 16 + c) * HW + y * W + x] = c9[((size_t)n * 64 + ch) * HW + y * W + x];
}

// Inclusive linear recurrence: h[t] = a[t]*h[t-1] + (1-a[t])*u[idx(t)], h[-1]=0.
// axis==3 -> scan along W; axis==2 -> scan along H. revU reverses u's index only.
__global__ void scan_kernel(const float* __restrict__ a, int aC, int aOff,
                            const float* __restrict__ u, float* __restrict__ out,
                            int N, int H, int W, int axis, int revU, int totalLines)
{
    int t = blockIdx.x * blockDim.x + threadIdx.x;
    if (t >= totalLines) return;
    int L    = (axis == 3) ? W : H;
    int step = (axis == 3) ? 1 : W;
    int perp = (axis == 3) ? H : W;
    int j = t % perp;
    int c = (t / perp) % 16;
    int n = t / (perp * 16);
    size_t lineOff = (axis == 3) ? (size_t)j * W : (size_t)j;
    size_t HW = (size_t)H * W;
    const float* ap = a  + ((size_t)n * aC + aOff + c) * HW + lineOff;
    const float* up = u  + ((size_t)n * 16 + c) * HW + lineOff;
    float*       op = out + ((size_t)n * 16 + c) * HW + lineOff;
    float h = 0.f;
    for (int i = 0; i < L; ++i) {
        float ai = ap[(size_t)i * step];
        int   ui = revU ? (L - 1 - i) : i;
        float uv = up[(size_t)ui * step];
        h = ai * h + (1.f - ai) * uv;
        op[(size_t)i * step] = h;
    }
}

__global__ void max4_kernel(const float* __restrict__ a, const float* __restrict__ b,
                            const float* __restrict__ c, const float* __restrict__ d,
                            float* __restrict__ o, int total)
{
    int t = blockIdx.x * blockDim.x + threadIdx.x;
    if (t >= total) return;
    o[t] = fmaxf(fmaxf(a[t], b[t]), fmaxf(c[t], d[t]));
}

// ---------------------------------------------------------------------------
// Host orchestration
// ---------------------------------------------------------------------------
#define ELT_LAUNCH_S(total, st) dim3(((total) + 255) / 256), dim3(256), 0, st
#define ELT_LAUNCH(total) ELT_LAUNCH_S(total, stream)

static void launch_conv(const float* in, const float* w, const float* b, float* out,
                        int N, int Cin, int H, int W, int Cout, int KW, int act,
                        _Float16* padbuf, hipStream_t st)
{
    const int P = KW / 2;
    {   // pad + convert input to f16 once
        int tot = N * Cin * (H + 2 * P) * (W + 2 * P);
        pad_f16_kernel<<<ELT_LAUNCH_S(tot, st)>>>(in, padbuf, N, Cin, H, W, P, tot);
    }

#define LCONV(KW_, ACT_, CIN_, COT_)                                                     \
    do {                                                                                 \
        constexpr int Ktot_ = (CIN_) * (KW_) * (KW_);                                    \
        constexpr int Kpad_ = ((Ktot_ + 31) / 32) * 32;                                  \
        int numTiles = (N * H * W) / 16;                                                 \
        dim3 grid((numTiles + 3) / 4, (Cout + 16 * (COT_) - 1) / (16 * (COT_)));         \
        size_t sh = (size_t)Kpad_ * 16 * (COT_) * sizeof(_Float16);                      \
        conv_wmma_kernel<KW_, ACT_, CIN_, COT_><<<grid, dim3(128), sh, st>>>(            \
            padbuf, w, b, out, N, H, W, Cout, numTiles);                                 \
    } while (0)

    if (KW == 5)                    LCONV(5, 1, 3, 1);      // c2: 3->16, 5x5, relu
    else if (act == 0 && Cin == 3)  LCONV(3, 0, 3, 1);      // s1: 3->3
    else if (act == 0)              LCONV(3, 0, 12, 1);     // mc: 12->16
    else if (act == 2)              LCONV(3, 2, 32, 2);     // c9: 32->64, tanh
    else if (Cin == 16)             LCONV(3, 1, 16, 2);     // c3: 16->32, c10: 16->64
    else if (Cin == 32)             LCONV(3, 1, 32, 2);     // c4,c5: 32->32, c6: 32->64
    else if (Cin == 96)             LCONV(3, 1, 96, 2);     // c7: 96->32
    else if (Cout >= 32)            LCONV(3, 1, 64, 2);     // c6s: 64->64
    else                            LCONV(3, 1, 64, 1);     // c8: 64->16, c11: 64->3
#undef LCONV
}

extern "C" void kernel_launch(void* const* d_in, const int* in_sizes, int n_in,
                              void* d_out, int out_size, void* d_ws, size_t ws_size,
                              hipStream_t stream) {
    const float* x    = (const float*)d_in[0];
    const float* w_s1 = (const float*)d_in[1];  const float* b_s1 = (const float*)d_in[2];
    const float* w_mc = (const float*)d_in[3];  const float* b_mc = (const float*)d_in[4];
    const float* w2   = (const float*)d_in[5];  const float* b2   = (const float*)d_in[6];
    const float* w3   = (const float*)d_in[7];  const float* b3   = (const float*)d_in[8];
    const float* w4   = (const float*)d_in[9];  const float* b4   = (const float*)d_in[10];
    const float* w5   = (const float*)d_in[11]; const float* b5   = (const float*)d_in[12];
    const float* w6   = (const float*)d_in[13]; const float* b6   = (const float*)d_in[14];
    const float* w6s  = (const float*)d_in[15]; const float* b6s  = (const float*)d_in[16];
    const float* w7   = (const float*)d_in[17]; const float* b7   = (const float*)d_in[18];
    const float* w8   = (const float*)d_in[19]; const float* b8   = (const float*)d_in[20];
    const float* w9   = (const float*)d_in[21]; const float* b9   = (const float*)d_in[22];
    const float* w10  = (const float*)d_in[23]; const float* b10  = (const float*)d_in[24];
    const float* w11  = (const float*)d_in[25]; const float* b11  = (const float*)d_in[26];

    const int N = 2, H = 512, W = 512;
    const int S = H * W;

    size_t off = 0;
    auto alloc = [&](size_t elems) -> float* {
        float* p = (float*)((char*)d_ws + off);
        off += elems * sizeof(float);
        off = (off + 255) & ~(size_t)255;
        return p;
    };

    // Shared padded-f16 scratch, sized for largest conv input: (N,64,514,514) halfs.
    size_t padHalfs = (size_t)N * 64 * 514 * 514;
    _Float16* padbuf = (_Float16*)alloc((padHalfs + 1) / 2);

    // ---- multi-scale branch ----
    float* s1 = alloc((size_t)N * 3 * S);
    launch_conv(x, w_s1, b_s1, s1, N, 3, 512, 512, 3, 3, 0, padbuf, stream);

    float* s2 = alloc((size_t)N * 3 * S / 4);
    { int tot = N * 3 * 256 * 256; maxpool_kernel<<<ELT_LAUNCH(tot)>>>(s1, s2, N, 3, 512, 512, tot); }
    float* s3 = alloc((size_t)N * 3 * S / 16);
    { int tot = N * 3 * 128 * 128; maxpool_kernel<<<ELT_LAUNCH(tot)>>>(s2, s3, N, 3, 256, 256, tot); }
    float* s4 = alloc((size_t)N * 3 * S / 64);
    { int tot = N * 3 * 64 * 64;   maxpool_kernel<<<ELT_LAUNCH(tot)>>>(s3, s4, N, 3, 128, 128, tot); }

    float* multi = alloc((size_t)N * 12 * S);
    { int tot = N * 3 * S; copy_ch_kernel<<<ELT_LAUNCH(tot)>>>(s1, 3, 0, multi, 12, 0, N, 3, S, tot); }
    { int tot = N * 3 * S; upsample_kernel<<<ELT_LAUNCH(tot)>>>(s2, 3, 0, multi, 12, 3, N, 3, 256, 256, 2, tot); }
    { int tot = N * 3 * S; upsample_kernel<<<ELT_LAUNCH(tot)>>>(s3, 3, 0, multi, 12, 6, N, 3, 128, 128, 4, tot); }
    { int tot = N * 3 * S; upsample_kernel<<<ELT_LAUNCH(tot)>>>(s4, 3, 0, multi, 12, 9, N, 3, 64, 64, 8, tot); }

    float* ms = alloc((size_t)N * 16 * S);
    launch_conv(multi, w_mc, b_mc, ms, N, 12, 512, 512, 16, 3, 0, padbuf, stream);

    // ---- encoder ----
    float* c2 = alloc((size_t)N * 16 * S);
    launch_conv(x, w2, b2, c2, N, 3, 512, 512, 16, 5, 1, padbuf, stream);
    float* c2p = alloc((size_t)N * 16 * S / 4);
    { int tot = N * 16 * 256 * 256; maxpool_kernel<<<ELT_LAUNCH(tot)>>>(c2, c2p, N, 16, 512, 512, tot); }

    float* c3 = alloc((size_t)N * 32 * S / 4);
    launch_conv(c2p, w3, b3, c3, N, 16, 256, 256, 32, 3, 1, padbuf, stream);
    float* c3p = alloc((size_t)N * 32 * S / 16);
    { int tot = N * 32 * 128 * 128; maxpool_kernel<<<ELT_LAUNCH(tot)>>>(c3, c3p, N, 32, 256, 256, tot); }

    float* c4 = alloc((size_t)N * 32 * S / 16);
    launch_conv(c3p, w4, b4, c4, N, 32, 128, 128, 32, 3, 1, padbuf, stream);
    float* c4p = alloc((size_t)N * 32 * S / 64);
    { int tot = N * 32 * 64 * 64; maxpool_kernel<<<ELT_LAUNCH(tot)>>>(c4, c4p, N, 32, 128, 128, tot); }

    float* c5 = alloc((size_t)N * 32 * S / 64);
    launch_conv(c4p, w5, b5, c5, N, 32, 64, 64, 32, 3, 1, padbuf, stream);
    float* c5p = alloc((size_t)N * 32 * S / 256);
    { int tot = N * 32 * 32 * 32; maxpool_kernel<<<ELT_LAUNCH(tot)>>>(c5, c5p, N, 32, 64, 64, tot); }

    float* c6 = alloc((size_t)N * 64 * S / 256);
    launch_conv(c5p, w6, b6, c6, N, 32, 32, 32, 64, 3, 1, padbuf, stream);

    // ---- decoder ----
    float* upc6 = alloc((size_t)N * 64 * S / 64);
    { int tot = N * 64 * 64 * 64; upsample_kernel<<<ELT_LAUNCH(tot)>>>(c6, 64, 0, upc6, 64, 0, N, 64, 32, 32, 2, tot); }
    float* c6s = alloc((size_t)N * 64 * S / 64);
    launch_conv(upc6, w6s, b6s, c6s, N, 64, 64, 64, 64, 3, 1, padbuf, stream);

    float* cat7 = alloc((size_t)N * 96 * S / 16);
    { int tot = N * 64 * 128 * 128; upsample_kernel<<<ELT_LAUNCH(tot)>>>(c6s, 64, 0, cat7, 96, 0, N, 64, 64, 64, 2, tot); }
    { int tot = N * 32 * 128 * 128; copy_ch_kernel<<<ELT_LAUNCH(tot)>>>(c4, 32, 0, cat7, 96, 64, N, 32, 128 * 128, tot); }
    float* c7 = alloc((size_t)N * 32 * S / 16);
    launch_conv(cat7, w7, b7, c7, N, 96, 128, 128, 32, 3, 1, padbuf, stream);

    float* cat8 = alloc((size_t)N * 64 * S / 4);
    { int tot = N * 32 * 256 * 256; upsample_kernel<<<ELT_LAUNCH(tot)>>>(c7, 32, 0, cat8, 64, 0, N, 32, 128, 128, 2, tot); }
    { int tot = N * 32 * 256 * 256; copy_ch_kernel<<<ELT_LAUNCH(tot)>>>(c3, 32, 0, cat8, 64, 32, N, 32, 256 * 256, tot); }
    float* c8 = alloc((size_t)N * 16 * S / 4);
    launch_conv(cat8, w8, b8, c8, N, 64, 256, 256, 16, 3, 1, padbuf, stream);

    float* cat9 = alloc((size_t)N * 32 * S);
    { int tot = N * 16 * S; upsample_kernel<<<ELT_LAUNCH(tot)>>>(c8, 16, 0, cat9, 32, 0, N, 16, 256, 256, 2, tot); }
    { int tot = N * 16 * S; copy_ch_kernel<<<ELT_LAUNCH(tot)>>>(c2, 16, 0, cat9, 32, 16, N, 16, S, tot); }

    float* c9 = alloc((size_t)N * 64 * S);
    launch_conv(cat9, w9, b9, c9, N, 32, 512, 512, 64, 3, 2, padbuf, stream);   // tanh

    // ---- gated linear scans ----
    float* gv = alloc((size_t)N * 16 * S);
    { int tot = N * 16 * S; gatev_kernel<<<ELT_LAUNCH(tot)>>>(c9, gv, N, H, W, tot); }

    float* h1 = alloc((size_t)N * 16 * S);
    float* h2 = alloc((size_t)N * 16 * S);
    float* h3 = alloc((size_t)N * 16 * S);
    float* h4 = alloc((size_t)N * 16 * S);
    float* h5 = alloc((size_t)N * 16 * S);
    float* h6 = alloc((size_t)N * 16 * S);
    float* h7 = alloc((size_t)N * 16 * S);
    float* h8 = alloc((size_t)N * 16 * S);

    int lines = N * 16 * 512;
    scan_kernel<<<ELT_LAUNCH(lines)>>>(c9, 64, 0,  ms, h1, N, H, W, 3, 0, lines);  // gx1
    scan_kernel<<<ELT_LAUNCH(lines)>>>(c9, 64, 0,  ms, h2, N, H, W, 3, 1, lines);
    scan_kernel<<<ELT_LAUNCH(lines)>>>(gv, 16, 0,  ms, h3, N, H, W, 2, 0, lines);  // gate_v
    scan_kernel<<<ELT_LAUNCH(lines)>>>(gv, 16, 0,  ms, h4, N, H, W, 2, 1, lines);
    scan_kernel<<<ELT_LAUNCH(lines)>>>(c9, 64, 32, h1, h5, N, H, W, 3, 0, lines);  // gx2
    scan_kernel<<<ELT_LAUNCH(lines)>>>(c9, 64, 32, h2, h6, N, H, W, 3, 1, lines);
    scan_kernel<<<ELT_LAUNCH(lines)>>>(c9, 64, 48, h3, h7, N, H, W, 2, 0, lines);  // gy2
    scan_kernel<<<ELT_LAUNCH(lines)>>>(c9, 64, 48, h4, h8, N, H, W, 2, 1, lines);

    float* emax = ms;   // ms is dead after h1..h4
    { int tot = N * 16 * S; max4_kernel<<<ELT_LAUNCH(tot)>>>(h5, h6, h7, h8, emax, tot); }

    // ---- head ----
    float* c10 = c9;    // c9 gates are dead after h5..h8
    launch_conv(emax, w10, b10, c10, N, 16, 512, 512, 64, 3, 1, padbuf, stream);
    launch_conv(c10, w11, b11, (float*)d_out, N, 64, 512, 512, 3, 3, 1, padbuf, stream);

    (void)in_sizes; (void)n_in; (void)out_size; (void)ws_size;
}
